// VAR_56040733278953
// MI455X (gfx1250) — matrix operand order repaired and checked by reading, not compiled, not run on hardware
//
#include <hip/hip_runtime.h>

// CDNA5 / gfx1250 — wave32, WMMA bf16 16x16x32, f32 accumulate.
// Strategy: one-time f32->bf16 conversion of W (fits in 192MB L2 as bf16),
// then 12 sequential bf16 GEMM steps with zero conversions in the hot loop.
typedef __attribute__((ext_vector_type(4)))  __bf16 v4bf;
typedef __attribute__((ext_vector_type(8)))  __bf16 v8bf;
typedef __attribute__((ext_vector_type(16))) __bf16 v16bf;
typedef __attribute__((ext_vector_type(8)))  float  v8f;

#define NN 2048   // nodes
#define BB 128    // batch
#define PP 12     // AR order / steps

// Convert W [12][2048][2048] f32 -> bf16, 4 elements per thread (b128 in, b64 out).
__global__ __launch_bounds__(256) void var_prep_w(const float* __restrict__ W,
                                                  __bf16* __restrict__ wbf) {
    size_t i4 = ((size_t)blockIdx.x * 256 + threadIdx.x) * 4;
    if (i4 >= (size_t)PP * NN * NN) return;
    const float4 v = *(const float4*)(W + i4);
    v4bf o;
    o[0] = (__bf16)v.x; o[1] = (__bf16)v.y; o[2] = (__bf16)v.z; o[3] = (__bf16)v.w;
    *(v4bf*)(wbf + i4) = o;
}

// Transpose+convert x[B][N][1][P] (lag fastest) -> xwbf[B][P][N] bf16.
__global__ __launch_bounds__(256) void var_prep_x(const float* __restrict__ x,
                                                  __bf16* __restrict__ xwbf) {
    size_t idx = (size_t)blockIdx.x * blockDim.x + threadIdx.x;
    if (idx >= (size_t)BB * PP * NN) return;
    int n = (int)(idx % NN);
    int l = (int)((idx / NN) % PP);
    int b = (int)(idx / ((size_t)NN * PP));
    xwbf[idx] = (__bf16)x[((size_t)b * NN + n) * PP + l];
}

// One AR step: y_t = Wmat^T * window, M=128, N=2048, K=12*2048, all bf16 in.
// Window row i at step t is xwbf[b][t+i] for i < 12-t, else ybuf[i-(12-t)][b].
// grid = 128 blocks (8 M-tiles x 16 N-groups), 256 threads = 8 waves,
// each wave computes one 16x16 f32 tile of y_t.
__global__ __launch_bounds__(256) void var_step(const __bf16* __restrict__ wbf,
                                                const __bf16* __restrict__ xwbf,
                                                __bf16* __restrict__ ybuf,
                                                float* __restrict__ out,
                                                int t) {
    const int lane   = threadIdx.x & 31;
    const int wave   = threadIdx.x >> 5;
    const int mtile  = blockIdx.x & 7;          // 8 tiles over B=128
    const int ngroup = blockIdx.x >> 3;         // 16 groups of 8 N-tiles
    const int ntile  = (ngroup << 3) | wave;    // 128 tiles over N=2048

    const int half = lane >> 4;                 // lane-half (K split)
    const int mr   = lane & 15;                 // row-in-tile (A) / col-in-tile (B)

    const int b = (mtile << 4) + mr;            // A: batch row
    const int j = (ntile << 4) + mr;            // B: output node column

    // A k order used here (same as B, not AMD's table order): K chunk [h*16, h*16+16), contiguous (two 16B loads)
    const int ka0 = half * 16;
    const int ka1 = half * 16 + 8;
    // B k order used here (same as A, not AMD's table order): K chunk [h*16, h*16+16), contiguous (32B)
    const int kb0 = half * 16;

    v8f acc = {};

    for (int i = 0; i < PP; ++i) {
        const __bf16* srow = (i < PP - t)
            ? (xwbf + ((size_t)b * PP + (t + i)) * NN)                 // original lags
            : (ybuf + ((size_t)(i - (PP - t)) * BB + b) * NN);         // prior outputs
        // Wmat[(i*NN + k), j] = W[i][j][k]: contiguous in k for fixed j
        const __bf16* wrow = wbf + ((size_t)i * NN + j) * NN;

#pragma unroll 2
        for (int kk = 0; kk < NN; kk += 32) {
            v8bf a0 = *(const v8bf*)(srow + kk + ka0);   // 16B aligned
            v8bf a1 = *(const v8bf*)(srow + kk + ka1);   // 16B aligned
            v16bf av = __builtin_shufflevector(a0, a1, 0, 1, 2, 3, 4, 5, 6, 7,
                                                       8, 9, 10, 11, 12, 13, 14, 15);
            v16bf bv = *(const v16bf*)(wrow + kk + kb0); // 32B aligned

            // D = A*B + C ; emits v_wmma_f32_16x16x32_bf16
            acc = __builtin_amdgcn_wmma_f32_16x16x32_bf16(
                /*neg_a=*/false, av, /*neg_b=*/false, bv,
                /*c_mod=*/(short)0, acc, /*reuse_a=*/false, /*reuse_b=*/false);
        }
    }

    // f32 16x16 C/D layout: lane col = mr, rows r + 8*half in acc[0..7]
    const int ncol = (ntile << 4) + mr;
#pragma unroll
    for (int r = 0; r < 8; ++r) {
        const int brow = (mtile << 4) + (half << 3) + r;
        const float v = acc[r];
        ybuf[((size_t)t * BB + brow) * NN + ncol] = (__bf16)v;  // feed recurrence
        out[((size_t)brow * NN + ncol) * PP + t] = v;           // [B][N][12] f32
    }
}

extern "C" void kernel_launch(void* const* d_in, const int* in_sizes, int n_in,
                              void* d_out, int out_size, void* d_ws, size_t ws_size,
                              hipStream_t stream) {
    (void)in_sizes; (void)n_in; (void)out_size; (void)ws_size;
    const float* x = (const float*)d_in[0];   // [128][2048][1][12] f32
    const float* W = (const float*)d_in[1];   // [12][2048][2048]   f32
    float* out = (float*)d_out;               // [128][2048][12]    f32

    // Workspace layout (bf16): W copy (100.7MB) | xw (6.3MB) | ybuf (6.3MB)
    __bf16* wbf  = (__bf16*)d_ws;
    __bf16* xwbf = wbf  + (size_t)PP * NN * NN;
    __bf16* ybuf = xwbf + (size_t)BB * PP * NN;

    const size_t wquads = (size_t)PP * NN * NN / 4;
    var_prep_w<<<(unsigned)((wquads + 255) / 256), 256, 0, stream>>>(W, wbf);
    const int xtotal = BB * PP * NN;
    var_prep_x<<<(xtotal + 255) / 256, 256, 0, stream>>>(x, xwbf);
    for (int t = 0; t < PP; ++t)
        var_step<<<128, 256, 0, stream>>>(wbf, xwbf, ybuf, out, t);
}
